// Attention_12137577578669
// MI455X (gfx1250) — matrix operand order repaired and checked by reading, not compiled, not run on hardware
//
#include <hip/hip_runtime.h>

typedef __attribute__((ext_vector_type(16))) _Float16 v16h;
typedef __attribute__((ext_vector_type(2)))  __fp16   h2;   // matches cvt_pkrtz return
typedef __attribute__((ext_vector_type(8)))  float    v8f;

constexpr int Bsz = 2, Lq = 4096, Sk = 4096, Hn = 8, EE = 64;
constexpr int BM = 128;    // Q rows per block
constexpr int BN = 64;     // keys per iteration
constexpr int NW = 8;      // waves per block
constexpr int NT = 256;    // threads per block
constexpr int KP = 72;     // padded LDS row stride (halves): 144B = 9*16B

// ---- DPP16 butterfly max within each 16-lane half ----
template <int CTRL>
__device__ __forceinline__ float dppf(float x) {
  int r = __builtin_amdgcn_update_dpp(0, __builtin_bit_cast(int, x),
                                      CTRL, 0xF, 0xF, true);
  return __builtin_bit_cast(float, r);
}
__device__ __forceinline__ float rowmax16(float x) {
  x = fmaxf(x, dppf<0xB1>(x));    // quad_perm(1,0,3,2)  : xor 1
  x = fmaxf(x, dppf<0x4E>(x));    // quad_perm(2,3,0,1)  : xor 2
  x = fmaxf(x, dppf<0x141>(x));   // row_half_mirror
  x = fmaxf(x, dppf<0x140>(x));   // row_mirror
  return x;
}
__device__ __forceinline__ float xhalfmax(float x) {
  // combine the two 16-lane halves: ds_swizzle xor-mask 16 (SWAPX16)
  int y = __builtin_amdgcn_ds_swizzle(__builtin_bit_cast(int, x), 0x401F);
  return fmaxf(x, __builtin_bit_cast(float, y));
}

__global__ __launch_bounds__(NT) void fa_wmma_kernel(
    const float* __restrict__ Q, const float* __restrict__ K,
    const float* __restrict__ V, float* __restrict__ O) {
  __shared__ __align__(16) _Float16 sK[BN * KP];          // [key][e], padded rows
  __shared__ __align__(16) _Float16 sVt[EE * KP];         // [d][key], padded rows
  __shared__ __align__(16) _Float16 sP[NW * 16 * KP];     // per-wave P, padded rows

  const int tid  = threadIdx.x;
  const int wave = tid >> 5;
  const int lane = tid & 31;
  const int l16  = lane & 15;
  const int h2i  = lane >> 4;

  const int qtile = blockIdx.x;
  const int h     = blockIdx.y;
  const int b     = blockIdx.z;

  const int    row_stride = Hn * EE;
  const size_t qbase = ((size_t)(b * Lq) * Hn + h) * EE;
  const size_t kbase = ((size_t)(b * Sk) * Hn + h) * EE;

  // ---- Q into A-fragment layout, pre-scaled by log2(e)/sqrt(E) (exp2 domain) ----
  const float scale = 0.125f * 1.44269504088896f;
  const int qrow = qtile * BM + wave * 16 + l16;
  const float* qp = Q + qbase + (size_t)qrow * row_stride;
  v16h qa[2];
  #pragma unroll
  for (int f = 0; f < 2; ++f) {
    const int e0 = 32 * f + 16 * h2i;     // elements 0..7  -> e0..e0+7
    const int e1 = e0 + 8;                // elements 8..15 -> e1..e1+7
    union { v16h v; h2 hp[8]; } u;
    #pragma unroll
    for (int i = 0; i < 4; ++i) {
      u.hp[i]     = __builtin_amdgcn_cvt_pkrtz(qp[e0 + 2*i] * scale, qp[e0 + 2*i + 1] * scale);
      u.hp[i + 4] = __builtin_amdgcn_cvt_pkrtz(qp[e1 + 2*i] * scale, qp[e1 + 2*i + 1] * scale);
    }
    qa[f] = u.v;
  }

  // ones B-fragment: row-sum of P via WMMA (l becomes a 5th accumulator column)
  v16h ones;
  #pragma unroll
  for (int j = 0; j < 16; ++j) ones[j] = (_Float16)1.0f;

  // ---- online-softmax state + output accumulators ----
  float mstripe = -3.0e38f;            // wave-uniform upper bound for all 16 rows
  v8f lacc = {};                       // row-sums, broadcast across 16 lanes
  v8f acc[4];
  #pragma unroll
  for (int t = 0; t < 4; ++t) { v8f z = {}; acc[t] = z; }

  // cooperative-load assignments
  const int k_row  = tid >> 2;           // 0..63
  const int k_eoff = (tid & 3) * 16;     // 0,16,32,48
  const int v_pair = tid >> 3;           // 0..31  -> keys 2p, 2p+1
  const int v_d0   = (tid & 7) * 8;      // 8 d-values per thread

  _Float16* pw = &sP[wave * 16 * KP];

  for (int s0 = 0; s0 < Sk; s0 += BN) {
    // ---- cooperative tile load: K row-major, V transposed, f32->f16 packed ----
    const float* kp  = K + kbase + (size_t)(s0 + k_row) * row_stride + k_eoff;
    const float* vp0 = V + kbase + (size_t)(s0 + 2 * v_pair) * row_stride + v_d0;
    const float* vp1 = vp0 + row_stride;
    {
      uint4* dstK = (uint4*)&sK[k_row * KP + k_eoff];
      #pragma unroll
      for (int i = 0; i < 2; ++i) {               // two b128 stores (8 halves each)
        float4 a = ((const float4*)kp)[2 * i];
        float4 c = ((const float4*)kp)[2 * i + 1];
        union { h2 hh[4]; uint4 u; } pk;
        pk.hh[0] = __builtin_amdgcn_cvt_pkrtz(a.x, a.y);
        pk.hh[1] = __builtin_amdgcn_cvt_pkrtz(a.z, a.w);
        pk.hh[2] = __builtin_amdgcn_cvt_pkrtz(c.x, c.y);
        pk.hh[3] = __builtin_amdgcn_cvt_pkrtz(c.z, c.w);
        dstK[i] = pk.u;
      }
      float4 va0 = ((const float4*)vp0)[0], va1 = ((const float4*)vp0)[1];
      float4 vb0 = ((const float4*)vp1)[0], vb1 = ((const float4*)vp1)[1];
      const float* fa = (const float*)&va0;
      const float* fb = (const float*)&vb0;
      #pragma unroll
      for (int d = 0; d < 8; ++d) {
        float xa = (d < 4) ? fa[d] : ((const float*)&va1)[d - 4];
        float xb = (d < 4) ? fb[d] : ((const float*)&vb1)[d - 4];
        *(h2*)&sVt[(v_d0 + d) * KP + 2 * v_pair] = __builtin_amdgcn_cvt_pkrtz(xa, xb);
      }
    }
    if (s0 + BN < Sk) {                          // gfx1250 global_prefetch_b8
      __builtin_prefetch(kp + (size_t)BN * row_stride, 0, 1);
      __builtin_prefetch(vp0 + (size_t)BN * row_stride, 0, 1);
    }
    __syncthreads();

    // ---- scores = Q * K^T (pre-scaled), permuted key order n = 2*l16 + (t&1) + 32*(t>>1) ----
    v8f sc[4];
    #pragma unroll
    for (int t = 0; t < 4; ++t) {
      const int key = 2 * l16 + (t & 1) + 32 * (t >> 1);   // physical key in tile
      const _Float16* kr = &sK[key * KP];
      v16h b0 = *(const v16h*)(kr + 16 * h2i);       // e = 0..31 block
      v16h b1 = *(const v16h*)(kr + 32 + 16 * h2i);  // e = 32..63 block
      v8f c = {};
      c = __builtin_amdgcn_wmma_f32_16x16x32_f16(false, qa[0], false, b0,
                                                 (short)0, c, false, false);
      c = __builtin_amdgcn_wmma_f32_16x16x32_f16(false, qa[1], false, b1,
                                                 (short)0, c, false, false);
      sc[t] = c;
    }

    // ---- online softmax, base-2 domain, wave-uniform stripe max ----
    float mx = fmaxf(fmaxf(sc[0][0], sc[1][0]), fmaxf(sc[2][0], sc[3][0]));
    #pragma unroll
    for (int r = 1; r < 8; ++r)
      mx = fmaxf(mx, fmaxf(fmaxf(sc[0][r], sc[1][r]), fmaxf(sc[2][r], sc[3][r])));
    mx = rowmax16(mx);
    mx = xhalfmax(mx);                           // now uniform across the wave
    const float mnew  = fmaxf(mstripe, mx);
    const float alpha = __builtin_amdgcn_exp2f(mstripe - mnew);
    mstripe = mnew;

    #pragma unroll
    for (int r = 0; r < 8; ++r) {
      const float p0 = __builtin_amdgcn_exp2f(sc[0][r] - mnew);
      const float p1 = __builtin_amdgcn_exp2f(sc[1][r] - mnew);
      const float p2 = __builtin_amdgcn_exp2f(sc[2][r] - mnew);
      const float p3 = __builtin_amdgcn_exp2f(sc[3][r] - mnew);
      const int rr = (r + 8 * h2i) * KP;
      *(h2*)&pw[rr + 2 * l16]      = __builtin_amdgcn_cvt_pkrtz(p0, p1);  // n=2*l16, +1
      *(h2*)&pw[rr + 32 + 2 * l16] = __builtin_amdgcn_cvt_pkrtz(p2, p3);  // n=32+2*l16, +1
    }
    // rescale only when the running max actually moved (uniform scalar branch;
    // alpha == 1.0f exactly when mnew == mstripe_old)
    if (__builtin_amdgcn_readfirstlane(__builtin_bit_cast(int, alpha)) != 0x3f800000) {
      lacc *= alpha;
      acc[0] *= alpha; acc[1] *= alpha; acc[2] *= alpha; acc[3] *= alpha;
    }
    // same-wave LDS write->read of sP: DS ops are in-order per wave

    // ---- out += P * V ; row-sums += P * 1 (K-dim split 2 x 32) ----
    #pragma unroll
    for (int split = 0; split < 2; ++split) {
      union { v16h v; uint4 q[2]; } ua;
      const int pbase = l16 * KP + 32 * split + 16 * h2i;
      ua.q[0] = *(const uint4*)(pw + pbase);         // n = base .. base+7
      ua.q[1] = *(const uint4*)(pw + pbase + 8);     // n = base+8 .. +15
      lacc = __builtin_amdgcn_wmma_f32_16x16x32_f16(false, ua.v, false, ones,
                                                    (short)0, lacc, false, false);
      #pragma unroll
      for (int t = 0; t < 4; ++t) {
        const int dcol = t * 16 + l16;
        v16h vb = *(const v16h*)(&sVt[dcol * KP + 32 * split + 16 * h2i]);
        acc[t] = __builtin_amdgcn_wmma_f32_16x16x32_f16(false, ua.v, false, vb,
                                                        (short)0, acc[t],
                                                        false, false);
      }
    }
    __syncthreads();
  }

  // ---- normalize and store O [B,L,H,E] ----
  #pragma unroll
  for (int r = 0; r < 8; ++r) {
    const int row_g = qtile * BM + wave * 16 + r + 8 * h2i;
    const float inv = 1.0f / lacc[r];
    float* op = O + qbase + (size_t)row_g * row_stride;
    #pragma unroll
    for (int t = 0; t < 4; ++t)
      op[t * 16 + l16] = acc[t][r] * inv;
  }
}

extern "C" void kernel_launch(void* const* d_in, const int* in_sizes, int n_in,
                              void* d_out, int out_size, void* d_ws, size_t ws_size,
                              hipStream_t stream) {
  (void)in_sizes; (void)n_in; (void)d_ws; (void)ws_size; (void)out_size;
  const float* q = (const float*)d_in[0];
  const float* k = (const float*)d_in[1];
  const float* v = (const float*)d_in[2];
  float* o = (float*)d_out;
  dim3 grid(Lq / BM, Hn, Bsz);
  dim3 block(NT);
  fa_wmma_kernel<<<grid, block, 0, stream>>>(q, k, v, o);
}
